// GCNFFN_23167053595561
// MI455X (gfx1250) — matrix operand order repaired and checked by reading, not compiled, not run on hardware
//
#include <hip/hip_runtime.h>
#include <math.h>

#define D_FEAT 128
#define XS_STRIDE 136   // 128 bf16 + 8 pad -> 272 B row stride (bank-conflict free, 16B aligned)

typedef __attribute__((ext_vector_type(16))) __bf16          v16bf;
typedef __attribute__((ext_vector_type(8)))  float           v8f;

struct P32 { uint4 lo, hi; };   // 32 bytes == v16bf

static __device__ __forceinline__ unsigned short f2bf(float f) {
  unsigned int u = __float_as_uint(f);
  u += 0x7FFFu + ((u >> 16) & 1u);      // round-to-nearest-even
  return (unsigned short)(u >> 16);
}

static __device__ __forceinline__ float gelu_exact(float v) {
  return 0.5f * v * (1.0f + erff(v * 0.70710678118654752f));
}

static __device__ __forceinline__ void atomic_fadd(float* p, float v) {
  __hip_atomic_fetch_add(p, v, __ATOMIC_RELAXED, __HIP_MEMORY_SCOPE_AGENT);
}

// ---------------------------------------------------------------- utility
__global__ void zero_f32_k(float* __restrict__ p, int n) {
  int i = blockIdx.x * blockDim.x + threadIdx.x;
  if (i < n) p[i] = 0.0f;
}

// deg[dst] += 1 per edge (self loop added in dinv_k)
__global__ void deg_count_k(const int* __restrict__ ei, float* __restrict__ deg, int E) {
  int e = blockIdx.x * blockDim.x + threadIdx.x;
  if (e < E) atomic_fadd(&deg[ei[E + e]], 1.0f);
}

// deg -> dinv in place; deg+1 (self loop) is always > 0
__global__ void dinv_k(float* __restrict__ deg, int n) {
  int i = blockIdx.x * blockDim.x + threadIdx.x;
  if (i < n) deg[i] = rsqrtf(deg[i] + 1.0f);
}

// ---------------------------------------------------------------- WMMA GEMM
// out[n, j] = sum_k x[n,k] * W[k,j]      (bias handled post-aggregation)
// LDS layouts are chosen so WMMA operands load as contiguous ds_load_b128:
//   Xs[m][k]  node-major  -> A regs = 2x b128 per K-step
//   Wt[j][k]  transposed  -> B regs = 2x b128 per (K-step, J-tile)
__global__ __launch_bounds__(128)
void gemm_bf16_k(const float* __restrict__ xin, int xin_col,
                 const float* __restrict__ W, float* __restrict__ out, int Nn)
{
  __shared__ __align__(16) unsigned short Wt[D_FEAT * XS_STRIDE]; // ~34 KB
  __shared__ __align__(16) unsigned short Xs[64 * XS_STRIDE];     // ~17 KB

  const int tid = threadIdx.x;
  const int nb  = blockIdx.x * 64;

  // stage W transposed: coalesced global reads of W[k][j], scattered LDS writes
  for (int i = tid; i < D_FEAT * D_FEAT; i += 128) {
    int k = i >> 7, j = i & 127;
    Wt[j * XS_STRIDE + k] = f2bf(W[i]);
  }
  // stage X tile node-major
  for (int i = tid; i < 64 * D_FEAT; i += 128) {
    int k, m;
    if (xin_col) { k = i >> 6;  m = i & 63; }     // coalesced over feature rows
    else         { m = i >> 7;  k = i & 127; }    // coalesced over node rows
    int n = nb + m;
    float v = 0.0f;
    if (n < Nn)
      v = xin_col ? xin[(size_t)k * Nn + n] : xin[(size_t)n * D_FEAT + k];
    Xs[m * XS_STRIDE + k] = f2bf(v);
  }
  __syncthreads();

  const int lane = tid & 31;
  const int wave = tid >> 5;
  const int half = lane >> 4;   // lane group (0: lanes 0-15, 1: lanes 16-31)
  const int mloc = lane & 15;
  const int mA   = wave * 16 + mloc;   // node index within block for A rows

  v8f acc[8];
#pragma unroll
  for (int jt = 0; jt < 8; jt++)
    acc[jt] = (v8f){0.f, 0.f, 0.f, 0.f, 0.f, 0.f, 0.f, 0.f};

#pragma unroll
  for (int kt = 0; kt < 4; kt++) {           // K = 128 in chunks of 32
    // A 16x32 bf16: lane holds M=mloc; K runs {half*8..+7} and {16+half*8..+7}
    const unsigned short* ap = &Xs[mA * XS_STRIDE + kt * 32 + half * 8];
    P32 at;
    at.lo = *(const uint4*)(ap);        // 8 bf16, K = half*8 ..
    at.hi = *(const uint4*)(ap + 16);   // 8 bf16, K = 16 + half*8 ..
    v16bf a = __builtin_bit_cast(v16bf, at);

#pragma unroll
    for (int jt = 0; jt < 8; jt++) {         // 8 column tiles of 16 (J = 128)
      // B 32x16 bf16: lane holds N=mloc; K runs {half*8..+7} and {16+half*8..+7}
      const unsigned short* bp = &Wt[(jt * 16 + mloc) * XS_STRIDE + kt * 32 + half * 8];
      P32 bt;
      bt.lo = *(const uint4*)(bp);
      bt.hi = *(const uint4*)(bp + 16);
      v16bf b = __builtin_bit_cast(v16bf, bt);
      acc[jt] = __builtin_amdgcn_wmma_f32_16x16x32_bf16(
          false, a, false, b, (short)0, acc[jt], false, false);
    }
  }

  // C/D layout: VGPR r -> M = half*8 + r, N = mloc
#pragma unroll
  for (int jt = 0; jt < 8; jt++) {
#pragma unroll
    for (int r = 0; r < 8; r++) {
      int mg = nb + wave * 16 + half * 8 + r;
      if (mg < Nn) out[(size_t)mg * D_FEAT + jt * 16 + mloc] = acc[jt][r];
    }
  }
}

// ---------------------------------------------------------------- aggregation
// Initialize agg with the self-loop term: dinv[n]^2 * xw[n, :]
__global__ void self_loop_init_k(const float* __restrict__ xw,
                                 const float* __restrict__ dinv,
                                 float* __restrict__ agg, int total)
{
  int i = blockIdx.x * blockDim.x + threadIdx.x;
  if (i < total) {
    int n = i >> 7;
    float d = dinv[n];
    agg[i] = d * d * xw[i];
  }
}

// One wave per 4 edges; lane owns 4 consecutive feats (float4 gather + 4 f32 atomics).
// Next edge's source row is prefetched (global_prefetch_b8) to hide gather latency.
#define EPW 4
__global__ __launch_bounds__(256)
void scatter_k(const float* __restrict__ xw, const int* __restrict__ ei,
               const float* __restrict__ dinv, float* __restrict__ agg, int E)
{
  int wid  = blockIdx.x * 8 + (threadIdx.x >> 5);
  int lane = threadIdx.x & 31;
  int e0   = wid * EPW;
  if (e0 >= E) return;

#pragma unroll
  for (int t = 0; t < EPW; t++) {
    int eid = e0 + t;
    if (eid >= E) return;
    int src = ei[eid];
    int dst = ei[E + eid];
    if (t + 1 < EPW && eid + 1 < E) {
      int nsrc = ei[eid + 1];
      __builtin_prefetch(xw + (size_t)nsrc * D_FEAT + lane * 4, 0, 0);
    }
    float norm = dinv[src] * dinv[dst];
    const float4 v = *((const float4*)(xw + (size_t)src * D_FEAT) + lane);
    float* base = agg + (size_t)dst * D_FEAT + (lane * 4);
    atomic_fadd(base + 0, v.x * norm);
    atomic_fadd(base + 1, v.y * norm);
    atomic_fadd(base + 2, v.z * norm);
    atomic_fadd(base + 3, v.w * norm);
  }
}

// ---------------------------------------------------------------- MessageNorm + GELU
// out[n,:] = gelu(x + scale * ||x|| * (msg+b)/max(||msg+b||, eps)) (+ residual)
// One wave per node; shuffle reduction over 32 lanes (wave32).
__global__ __launch_bounds__(256)
void msgnorm_gelu_k(const float* __restrict__ xin, int xin_col,
                    const float* __restrict__ msg, const float* __restrict__ bias,
                    const float* __restrict__ scale, const float* __restrict__ resid,
                    float* __restrict__ out, int Nn)
{
  int node = blockIdx.x * 8 + (threadIdx.x >> 5);
  if (node >= Nn) return;
  int lane = threadIdx.x & 31;

  float m[4], x[4];
#pragma unroll
  for (int i = 0; i < 4; i++) {
    int j = lane * 4 + i;
    m[i] = msg[(size_t)node * D_FEAT + j] + bias[j];
    x[i] = xin_col ? xin[(size_t)j * Nn + node] : xin[(size_t)node * D_FEAT + j];
  }
  float ms = 0.f, xs = 0.f;
#pragma unroll
  for (int i = 0; i < 4; i++) { ms += m[i] * m[i]; xs += x[i] * x[i]; }
#pragma unroll
  for (int off = 16; off > 0; off >>= 1) {
    ms += __shfl_xor(ms, off, 32);
    xs += __shfl_xor(xs, off, 32);
  }
  float sc = scale[0] * sqrtf(xs) / fmaxf(sqrtf(ms), 1e-12f);
#pragma unroll
  for (int i = 0; i < 4; i++) {
    int j = lane * 4 + i;
    float g = gelu_exact(x[i] + sc * m[i]);
    if (resid) g += resid[(size_t)j * Nn + node];   // col-major X residual
    out[(size_t)node * D_FEAT + j] = g;
  }
}

// ---------------------------------------------------------------- GraphNorm
// Pass 1: per-column sum and sum-of-squares (var recovered algebraically).
__global__ __launch_bounds__(128)
void colstats_k(const float* __restrict__ h, float* __restrict__ colsum,
                float* __restrict__ colsq, int Nn, int nodesPerBlk)
{
  int j  = threadIdx.x;            // column 0..127
  int n0 = blockIdx.x * nodesPerBlk;
  float s = 0.f, q = 0.f;
  for (int r = 0; r < nodesPerBlk; r++) {
    int n = n0 + r;
    if (n < Nn) {
      float v = h[(size_t)n * D_FEAT + j];
      s += v; q += v * v;
    }
  }
  atomic_fadd(&colsum[j], s);
  atomic_fadd(&colsq[j], q);
}

// out = w*(h - mean*ms)*rsqrt(var+eps) + b  ==  a*h + c  with
// e = mean*ms ; var = E[h^2] - 2*e*mean + e^2 ; a = w*rsqrt(var+eps) ; c = b - a*e
__global__ void gn_finalize_k(const float* __restrict__ colsum, const float* __restrict__ colsq,
                              const float* __restrict__ w, const float* __restrict__ bias,
                              const float* __restrict__ ms,
                              float* __restrict__ aOut, float* __restrict__ cOut, int Nn)
{
  int j = threadIdx.x;
  float invN = 1.0f / (float)Nn;
  float mean = colsum[j] * invN;
  float e    = mean * ms[j];
  float var  = colsq[j] * invN - 2.0f * e * mean + e * e;
  float a    = w[j] * rsqrtf(var + 1e-5f);
  aOut[j] = a;
  cOut[j] = bias[j] - a * e;
}

__global__ void gn_apply_k(float* __restrict__ h, const float* __restrict__ a,
                           const float* __restrict__ c, int total)
{
  int i = blockIdx.x * blockDim.x + threadIdx.x;
  if (i < total) {
    int j = i & 127;
    h[i] = gelu_exact(a[j] * h[i] + c[j]);
  }
}

// ---------------------------------------------------------------- launcher
extern "C" void kernel_launch(void* const* d_in, const int* in_sizes, int n_in,
                              void* d_out, int out_size, void* d_ws, size_t ws_size,
                              hipStream_t stream)
{
  (void)n_in; (void)out_size; (void)ws_size;
  const float* X   = (const float*)d_in[0];   // [128, N] (column-major node features)
  const int*   ei  = (const int*)d_in[1];     // [2, E]
  const float* W1  = (const float*)d_in[2];
  const float* b1  = (const float*)d_in[3];
  const float* s1  = (const float*)d_in[4];
  const float* W2  = (const float*)d_in[5];
  const float* b2  = (const float*)d_in[6];
  const float* s2  = (const float*)d_in[7];
  const float* gw  = (const float*)d_in[8];
  const float* gb  = (const float*)d_in[9];
  const float* gms = (const float*)d_in[10];

  const int Nn = in_sizes[0] / D_FEAT;
  const int E  = in_sizes[1] / 2;
  const int ND = Nn * D_FEAT;

  // workspace carve-out (256 B aligned)
  char* ws = (char*)d_ws;
  size_t off = 0;
  auto take = [&](size_t bytes) { char* p = ws + off; off += (bytes + 255) & ~(size_t)255; return p; };
  float* dinv   = (float*)take((size_t)Nn * 4);          // deg -> dinv in place
  float* xw     = (float*)take((size_t)ND * 4);
  float* agg    = (float*)take((size_t)ND * 4);
  float* f1     = (float*)take((size_t)ND * 4);
  float* colsum = (float*)take(512 * 4);                  // colsum|colsq|a|c contiguous
  float* colsq  = colsum + 128;
  float* aCoef  = colsum + 256;
  float* cCoef  = colsum + 384;

  float* outF = (float*)d_out;

  // degrees -> dinv
  zero_f32_k<<<(Nn + 255) / 256, 256, 0, stream>>>(dinv, Nn);
  deg_count_k<<<(E + 255) / 256, 256, 0, stream>>>(ei, dinv, E);
  dinv_k<<<(Nn + 255) / 256, 256, 0, stream>>>(dinv, Nn);

  const int gemmBlocks    = (Nn + 63) / 64;
  const int scatterBlocks = (E + 8 * EPW - 1) / (8 * EPW);

  // ---- layer 1: x (col-major X) ----
  gemm_bf16_k<<<gemmBlocks, 128, 0, stream>>>(X, 1, W1, xw, Nn);
  self_loop_init_k<<<(ND + 255) / 256, 256, 0, stream>>>(xw, dinv, agg, ND);
  scatter_k<<<scatterBlocks, 256, 0, stream>>>(xw, ei, dinv, agg, E);
  msgnorm_gelu_k<<<(Nn + 7) / 8, 256, 0, stream>>>(X, 1, agg, b1, s1, nullptr, f1, Nn);

  // ---- layer 2: f1 (row-major), residual = X^T added into output ----
  gemm_bf16_k<<<gemmBlocks, 128, 0, stream>>>(f1, 0, W2, xw, Nn);
  self_loop_init_k<<<(ND + 255) / 256, 256, 0, stream>>>(xw, dinv, agg, ND);
  scatter_k<<<scatterBlocks, 256, 0, stream>>>(xw, ei, dinv, agg, E);
  msgnorm_gelu_k<<<(Nn + 7) / 8, 256, 0, stream>>>(f1, 0, agg, b2, s2, X, outF, Nn);

  // ---- GraphNorm + GELU, in place on d_out ----
  zero_f32_k<<<1, 256, 0, stream>>>(colsum, 256);   // zeros colsum + colsq
  const int npb = 512;
  colstats_k<<<(Nn + npb - 1) / npb, 128, 0, stream>>>(outF, colsum, colsq, Nn, npb);
  gn_finalize_k<<<1, 128, 0, stream>>>(colsum, colsq, gw, gb, gms, aCoef, cCoef, Nn);
  gn_apply_k<<<(ND + 255) / 256, 256, 0, stream>>>(outF, aCoef, cCoef, ND);
}
